// BooleanReservoir_798863917195
// MI455X (gfx1250) — matrix operand order repaired and checked by reading, not compiled, not run on hardware
//
#include <hip/hip_runtime.h>

// ---------------- problem constants (from reference) ----------------
#define N_NODES   50000
#define K_MAX     12
#define BATCH     64
#define STEPS     50
#define N_IN      16          // N_INPUTS * BITS
#define LUT_WORDS 128         // 4096 bits / 32
#define KPAD      50016       // N_NODES padded to multiple of 32 for WMMA K-loop
#define NPADCOL   16          // B matrix padded to 16 columns for WMMA N

typedef __attribute__((ext_vector_type(16))) _Float16 v16h;
typedef __attribute__((ext_vector_type(8)))  float    v8f;

struct alignas(16) U4u { unsigned x, y, z, w; };
struct alignas(16) U4i { int x, y, z, w; };
struct HV32 { U4u lo, hi; };   // 32 bytes -> bit_cast to v16h

// ---------------- workspace layout ----------------
constexpr size_t alignUp(size_t v) { return (v + 255) & ~size_t(255); }
constexpr size_t SZ_LUTBITS = (size_t)N_NODES * LUT_WORDS * 4;   // 25.6 MB
constexpr size_t SZ_STATE   = (size_t)N_NODES * 2 * 4;           // 400 KB (2 batch-halves)
constexpr size_t SZ_MASK    = (size_t)N_NODES * 4;
constexpr size_t SZ_XW      = (size_t)STEPS * 2 * N_IN * 4;
constexpr size_t SZ_AH      = (size_t)BATCH * KPAD * 2;          // 6.4 MB f16
constexpr size_t SZ_WP      = (size_t)NPADCOL * KPAD * 2;        // 1.6 MB f16 (zero-padded B)

constexpr size_t OFF_LUT = 0;
constexpr size_t OFF_STA = alignUp(OFF_LUT + SZ_LUTBITS);
constexpr size_t OFF_STB = alignUp(OFF_STA + SZ_STATE);
constexpr size_t OFF_MSK = alignUp(OFF_STB + SZ_STATE);
constexpr size_t OFF_XW  = alignUp(OFF_MSK + SZ_MASK);
constexpr size_t OFF_AH  = alignUp(OFF_XW  + SZ_XW);
constexpr size_t OFF_WP  = alignUp(OFF_AH  + SZ_AH);

__device__ __forceinline__ unsigned ballot32(bool p) {
#if __has_builtin(__builtin_amdgcn_ballot_w32)
    return __builtin_amdgcn_ballot_w32(p);
#else
    return (unsigned)__ballot((int)p);
#endif
}

// -------- pack 819 MB int32 LUT -> 25.6 MB bit LUT (one streaming pass) --------
__global__ void pack_lut_kernel(const int* __restrict__ lut, unsigned* __restrict__ lutbits) {
    unsigned gid = blockIdx.x * 256u + threadIdx.x;          // one entry per lane
    int v = lut[gid];
    unsigned m = ballot32(v != 0);                           // wave32: 32 entries -> 1 word
    if ((threadIdx.x & 31u) == 0u) lutbits[gid >> 5] = m;
}

__global__ void pack_mask_kernel(const int* __restrict__ amask, unsigned* __restrict__ mask12) {
    int n = blockIdx.x * 256 + threadIdx.x;
    if (n >= N_NODES) return;
    unsigned m = 0;
#pragma unroll
    for (int k = 0; k < K_MAX; ++k) m |= (amask[n * K_MAX + k] != 0 ? 1u : 0u) << k;
    mask12[n] = m;
}

__global__ void pack_init_kernel(const int* __restrict__ init, unsigned* __restrict__ st) {
    int n = blockIdx.x * 256 + threadIdx.x;
    if (n >= N_NODES) return;
    unsigned v = (init[n] != 0) ? 0xFFFFFFFFu : 0u;          // same bit for every batch
    st[n] = v;                 // batch-half 0 (batches 0..31)
    st[N_NODES + n] = v;       // batch-half 1 (batches 32..63)
}

// xw[(s*2+h)*N_IN + j] : bit b = x[batch 32h+b][s][j]
__global__ void pack_x_kernel(const int* __restrict__ x, unsigned* __restrict__ xw) {
    int o = blockIdx.x * 256 + threadIdx.x;
    if (o >= STEPS * 2 * N_IN) return;
    int j = o & (N_IN - 1);
    int h = (o >> 4) & 1;
    int s = o >> 5;
    unsigned wv = 0;
    for (int b = 0; b < 32; ++b) {
        int m = 32 * h + b;
        wv |= (x[(m * STEPS + s) * N_IN + j] != 0 ? 1u : 0u) << b;
    }
    xw[o] = wv;
}

// Zero-padded f16 B matrix: Wp[col][k], 16 columns (2 real), KPAD rows.
__global__ void pack_wp_kernel(const float* __restrict__ W, _Float16* __restrict__ Wp) {
    int i = blockIdx.x * 256 + threadIdx.x;
    if (i >= NPADCOL * KPAD) return;
    int col = i / KPAD, n = i % KPAD;
    float v = (col < 2 && n < N_NODES) ? W[col * N_NODES + n] : 0.0f;
    Wp[i] = (_Float16)v;
}

// XOR step inputs into the 16 input nodes (both batch halves), pre-gather.
__global__ void xor_inputs_kernel(const int* __restrict__ input_nodes,
                                  const unsigned* __restrict__ xw,
                                  unsigned* __restrict__ cur, int s) {
    int t = threadIdx.x;           // 32 threads: (h, j)
    int h = t >> 4, j = t & 15;
    int node = input_nodes[j];
    cur[h * N_NODES + node] ^= xw[(s * 2 + h) * N_IN + j];
}

// One reservoir step: lane = node, blockIdx.y = batch half (32 batches bit-parallel)
__global__ void step_kernel(const int* __restrict__ adj,
                            const unsigned* __restrict__ mask12,
                            const unsigned* __restrict__ lutbits,
                            const unsigned* __restrict__ src,
                            unsigned* __restrict__ dst) {
    int n = blockIdx.x * 256 + threadIdx.x;
    if (n >= N_NODES) return;
    int h = blockIdx.y;
    const unsigned* sp = src + h * N_NODES;

    const U4i* ar = reinterpret_cast<const U4i*>(adj + (size_t)n * K_MAX); // 48B, 16B aligned
    U4i a0 = ar[0], a1 = ar[1], a2 = ar[2];
    int nb[K_MAX] = {a0.x, a0.y, a0.z, a0.w, a1.x, a1.y, a1.z, a1.w, a2.x, a2.y, a2.z, a2.w};
    unsigned msk = mask12[n];

    unsigned w[K_MAX];
#pragma unroll
    for (int k = 0; k < K_MAX; ++k) {
        unsigned v = sp[nb[k]];                       // 4B gather: 32 batches at once (L2-resident)
        w[k] = ((msk >> k) & 1u) ? v : 0u;
    }

    const unsigned char* lrow = reinterpret_cast<const unsigned char*>(lutbits) + (size_t)n * 512;
    unsigned out = 0;
#pragma unroll
    for (int b = 0; b < 32; ++b) {                    // per-batch 12-bit LUT index
        unsigned idx = 0;
#pragma unroll
        for (int k = 0; k < K_MAX; ++k)
            idx = (idx << 1) | ((w[k] >> b) & 1u);    // k=0 -> MSB (matches 2^(K-1-k) powers)
        unsigned bit = (lrow[idx >> 3] >> (idx & 7u)) & 1u;
        out |= bit << b;
    }
    dst[h * N_NODES + n] = out;
}

// Unpack final bit states -> f16 activation matrix A[64][KPAD] (zero-padded K)
__global__ void unpack_states_kernel(const unsigned* __restrict__ fin, _Float16* __restrict__ A) {
    int n = blockIdx.x * 256 + threadIdx.x;
    if (n >= KPAD) return;
    unsigned w0 = 0, w1 = 0;
    if (n < N_NODES) { w0 = fin[n]; w1 = fin[N_NODES + n]; }
#pragma unroll
    for (int m = 0; m < BATCH; ++m) {
        unsigned w = (m < 32) ? w0 : w1;
        unsigned bit = (w >> (m & 31)) & 1u;
        A[(size_t)m * KPAD + n] = bit ? (_Float16)1.0f : (_Float16)0.0f;
    }
}

// out[64][2] = A(64 x K) * Wp + b  via v_wmma_f32_16x16x32_f16.
// 4 waves, one 16-row M-tile each; B pre-padded to 16 columns so the inner
// loop is 4 unconditional b128 loads + 1 WMMA (no EXEC divergence).
__global__ void proj_wmma_kernel(const _Float16* __restrict__ A,
                                 const _Float16* __restrict__ Wp,
                                 const float* __restrict__ bias,
                                 float* __restrict__ out) {
    int lane  = threadIdx.x;
    int mbase = blockIdx.x * 16;
    int hiH   = lane >> 4;                 // lane group
    int row   = mbase + (lane & 15);       // A: M = lane&15
    int aOff0 = hiH * 8;                   // A VGPR0-3: K = 0..7 (lo) / 8..15 (hi)
    int aOff1 = aOff0 + 16;                // A VGPR4-7: K = 16..23 / 24..31
    int col   = lane & 15;                 // B: N = lane&15
    int bOff  = hiH * 8;                   // B: K = 0..7,16..23 (lo lanes) / 8..15,24..31 (hi lanes)

    v8f c = {0.f, 0.f, 0.f, 0.f, 0.f, 0.f, 0.f, 0.f};
    const _Float16* pa = A  + (size_t)row * KPAD;
    const _Float16* pb = Wp + (size_t)col * KPAD;

    for (int kb = 0; kb < KPAD; kb += 32) {
        HV32 av, bv;
        av.lo = *reinterpret_cast<const U4u*>(pa + kb + aOff0);
        av.hi = *reinterpret_cast<const U4u*>(pa + kb + aOff1);
        bv.lo = *reinterpret_cast<const U4u*>(pb + kb + bOff);
        bv.hi = *reinterpret_cast<const U4u*>(pb + kb + bOff + 16);
        v16h af = __builtin_bit_cast(v16h, av);
        v16h bf = __builtin_bit_cast(v16h, bv);
        c = __builtin_amdgcn_wmma_f32_16x16x32_f16(false, af, false, bf, (short)0, c,
                                                   false, false);
    }

    if (col < 2) {
        float bb = bias[col];
#pragma unroll
        for (int r = 0; r < 8; ++r) {       // D VGPR r: M = r (lo lanes) / 8+r (hi lanes)
            int m = mbase + r + hiH * 8;
            out[m * 2 + col] = c[r] + bb;
        }
    }
}

// ---------------- launcher ----------------
extern "C" void kernel_launch(void* const* d_in, const int* in_sizes, int n_in,
                              void* d_out, int out_size, void* d_ws, size_t ws_size,
                              hipStream_t stream) {
    const int*   x       = (const int*)d_in[0];   // (64,50,2,8) bool->int32
    const int*   adj     = (const int*)d_in[1];   // (50000,12) int32
    const int*   amask   = (const int*)d_in[2];   // (50000,12) bool->int32
    const int*   lut     = (const int*)d_in[3];   // (50000,4096) bool->int32
    const int*   innodes = (const int*)d_in[4];   // (16,) int32
    const int*   init    = (const int*)d_in[5];   // (50000,) bool->int32
    const float* W       = (const float*)d_in[6]; // (2,50000) f32
    const float* bias    = (const float*)d_in[7]; // (2,) f32
    float*       out     = (float*)d_out;         // (64,2) f32

    char* ws = (char*)d_ws;
    unsigned*  lutbits = (unsigned*)(ws + OFF_LUT);
    unsigned*  stA     = (unsigned*)(ws + OFF_STA);
    unsigned*  stB     = (unsigned*)(ws + OFF_STB);
    unsigned*  mask12  = (unsigned*)(ws + OFF_MSK);
    unsigned*  xw      = (unsigned*)(ws + OFF_XW);
    _Float16*  Ah      = (_Float16*)(ws + OFF_AH);
    _Float16*  Wp      = (_Float16*)(ws + OFF_WP);

    // preprocessing (deterministic, every call)
    pack_lut_kernel <<<(N_NODES * 4096) / 256, 256, 0, stream>>>(lut, lutbits);
    pack_mask_kernel<<<(N_NODES + 255) / 256, 256, 0, stream>>>(amask, mask12);
    pack_init_kernel<<<(N_NODES + 255) / 256, 256, 0, stream>>>(init, stA);
    pack_x_kernel   <<<(STEPS * 2 * N_IN + 255) / 256, 256, 0, stream>>>(x, xw);
    pack_wp_kernel  <<<(NPADCOL * KPAD + 255) / 256, 256, 0, stream>>>(W, Wp);

    // 50 sequential reservoir steps (global sync via kernel boundaries)
    unsigned* cur = stA;
    unsigned* nxt = stB;
    dim3 sgrid((N_NODES + 255) / 256, 2);
    for (int s = 0; s < STEPS; ++s) {
        xor_inputs_kernel<<<1, 32, 0, stream>>>(innodes, xw, cur, s);
        step_kernel<<<sgrid, 256, 0, stream>>>(adj, mask12, lutbits, cur, nxt);
        unsigned* t = cur; cur = nxt; nxt = t;
    }

    // final projection with WMMA
    unpack_states_kernel<<<(KPAD + 255) / 256, 256, 0, stream>>>(cur, Ah);
    proj_wmma_kernel<<<4, 32, 0, stream>>>(Ah, Wp, bias, out);
}
